// GAT_skip_forward_15135464751860
// MI455X (gfx1250) — matrix operand order repaired and checked by reading, not compiled, not run on hardware
//
#include <hip/hip_runtime.h>
#include <hip/hip_bf16.h>
#include <cstdint>
#include <cstddef>

#define NNODES 20000
#define NEDGES 480000
#define ETOTAL (NEDGES + NNODES)
#define NFEAT  128
#define NHEAD  8
#define CCH    32
#define HCH    256
#define NCLS   10
#define NCONVS 3
#define EPSBN  1e-5f
#define SLOPE  0.2f

typedef __attribute__((ext_vector_type(16))) __bf16 v16bf;
typedef __attribute__((ext_vector_type(8)))  float  v8f;
typedef __attribute__((ext_vector_type(4)))  unsigned u32x4;

union Frag16 { v16bf v; u32x4 q[2]; };

// ---------------- device helpers ----------------
__device__ __forceinline__ unsigned short f2bf(float f) {
  unsigned u = __float_as_uint(f);
  u += 0x7FFFu + ((u >> 16) & 1u);               // round-to-nearest-even
  return (unsigned short)(u >> 16);
}
__device__ __forceinline__ unsigned fenc(float f) {   // order-preserving float->uint
  int b = __float_as_int(f);
  return (unsigned)(b ^ ((b >> 31) | 0x80000000));
}
__device__ __forceinline__ float fdec(unsigned u) {
  int b = (u & 0x80000000u) ? (int)(u ^ 0x80000000u) : ~(int)u;
  return __int_as_float(b);
}

// ---------------- elementwise utility kernels ----------------
__global__ void fill_f32_kernel(float* __restrict__ p, float v, int n) {
  int t = blockIdx.x * blockDim.x + threadIdx.x;
  if (t < n) p[t] = v;
}
__global__ void fill_u32_kernel(unsigned* __restrict__ p, unsigned v, int n) {
  int t = blockIdx.x * blockDim.x + threadIdx.x;
  if (t < n) p[t] = v;
}
__global__ void init_bias_kernel(float* __restrict__ out, const float* __restrict__ bias,
                                 int n, int f) {
  int t = blockIdx.x * blockDim.x + threadIdx.x;
  if (t < n) out[t] = bias[t % f];
}
__global__ void to_bf16_kernel(const float* __restrict__ x, unsigned short* __restrict__ o, int n) {
  int t = blockIdx.x * blockDim.x + threadIdx.x;
  if (t < n) o[t] = f2bf(x[t]);
}

// ---------------- BatchNorm stats: coalesced row sweep + f32 atomics ----------------
// blockDim = 256; thread handles column c = tid % f, rows strided across grid.
__global__ void bn_partial_kernel(const float* __restrict__ x, const float* __restrict__ res,
                                  float* __restrict__ sumb, float* __restrict__ sqb,
                                  int n, int f) {
  int tid = threadIdx.x;
  int c = tid % f;                 // f in {128, 256}; 256 % f == 0
  int rsub = tid / f;
  int rowsPer = 256 / f;
  float a = 0.f, b2 = 0.f;
  for (int r = blockIdx.x * rowsPer + rsub; r < n; r += gridDim.x * rowsPer) {
    float v = x[(size_t)r * f + c];
    if (res) v += res[(size_t)r * f + c];
    a += v; b2 += v * v;
  }
  atomicAdd(&sumb[c], a);
  atomicAdd(&sqb[c], b2);
}
__global__ void bn_finalize_kernel(const float* __restrict__ sumb, const float* __restrict__ sqb,
                                   float* __restrict__ mu, float* __restrict__ inv,
                                   int n, int f) {
  int c = blockIdx.x * blockDim.x + threadIdx.x;
  if (c >= f) return;
  float m = sumb[c] / (float)n;
  float v = sqb[c] / (float)n - m * m;   // biased variance (matches jnp.var)
  mu[c] = m;
  inv[c] = rsqrtf(v + EPSBN);
}

__global__ void bn_apply_kernel(const float* __restrict__ x, const float* __restrict__ res,
                                const float* __restrict__ mu, const float* __restrict__ inv,
                                const float* __restrict__ g, const float* __restrict__ b,
                                float* __restrict__ out, int n, int f, int do_elu) {
  int t = blockIdx.x * blockDim.x + threadIdx.x;
  if (t >= n) return;
  int c = t % f;
  float v = x[t];
  if (res) v += res[t];
  float y = (v - mu[c]) * inv[c] * g[c] + b[c];
  if (do_elu) y = (y > 0.f) ? y : expm1f(y);
  out[t] = y;
}

// ---------------- weight packing into B-fragment layout ----------------
// packed idx = ((kt*NT + nt)*32 + lane)*16 + e
//   lane<16 : value = W[kt*32 +     (e<8 ? e : e+8)][nt*16 + lane]
//   lane>=16: value = W[kt*32 + 8 + (e<8 ? e : e+8)][nt*16 + (lane-16)]
__global__ void pack_w_kernel(const float* __restrict__ w, unsigned short* __restrict__ wp,
                              int K, int Nout) {
  int total = K * Nout;
  int t = blockIdx.x * blockDim.x + threadIdx.x;
  if (t >= total) return;
  int e = t & 15;
  int l = (t >> 4) & 31;
  int tile = t >> 9;
  int NT = Nout >> 4;
  int nt = tile % NT;
  int kt = tile / NT;
  int k = kt * 32 + ((e & 8) ? 16 : 0) + ((l & 16) ? 8 : 0) + (e & 7);
  int n = nt * 16 + (l & 15);
  wp[t] = f2bf(w[(size_t)k * Nout + n]);
}

// ---------------- WMMA GEMM: out[M,Nout] = bf16(A[M,K]) @ W + bias ----------------
// 256 threads = 8 waves; wave -> (w&1)*16 rows, (w>>1)*64 cols.
// NCT (compile-time) column tiles per wave => branch-free k-loop, EXEC all-ones at
// every v_wmma. Software-pipelined (double-buffered) fragment loads.
template <int NCT>
__global__ __launch_bounds__(256)
void gemm_bf16_wmma_kernel(const unsigned short* __restrict__ A,
                           const unsigned short* __restrict__ Wp,
                           const float* __restrict__ bias,
                           float* __restrict__ out,
                           int M, int K, int Nout, int do_elu) {
  const int lane = threadIdx.x & 31;
  const int wave = threadIdx.x >> 5;
  const int mBase = blockIdx.y * 32 + (wave & 1) * 16;
  const int nBase = blockIdx.x * 256 + (wave >> 1) * 64;
  if (mBase >= M) return;
  if (nBase >= Nout) return;                       // wave-uniform in practice

  const int NT = Nout >> 4;
  const int KT = K >> 5;
  const int halfSel = (lane >> 4) & 1;
  const unsigned short* arow =
      A + (size_t)(mBase + (lane & 15)) * K + (halfSel ? 8 : 0);
  const unsigned short* bbase =
      Wp + (size_t)(nBase >> 4) * 512 + (size_t)lane * 16;
  const size_t kStrideB = (size_t)NT * 512;        // packed elements per k-step

  const v8f z8 = {0.f, 0.f, 0.f, 0.f, 0.f, 0.f, 0.f, 0.f};
  v8f acc[NCT];
  Frag16 aCur, bCur[NCT];

  // prologue: kt = 0 fragments
  aCur.q[0] = *(const u32x4*)(arow);
  aCur.q[1] = *(const u32x4*)(arow + 16);
#pragma unroll
  for (int ct = 0; ct < NCT; ++ct) {
    const u32x4* bp = (const u32x4*)(bbase + (size_t)ct * 512);
    bCur[ct].q[0] = bp[0];
    bCur[ct].q[1] = bp[1];
    acc[ct] = z8;
  }

  for (int kt = 0; kt < KT; ++kt) {
    const int ktn = (kt + 1 < KT) ? (kt + 1) : kt;   // scalar select, no branch
    Frag16 aNxt, bNxt[NCT];
    const unsigned short* apn = arow + ktn * 32;
    aNxt.q[0] = *(const u32x4*)(apn);
    aNxt.q[1] = *(const u32x4*)(apn + 16);
    const unsigned short* bpn = bbase + (size_t)ktn * kStrideB;
#pragma unroll
    for (int ct = 0; ct < NCT; ++ct) {
      const u32x4* bp = (const u32x4*)(bpn + (size_t)ct * 512);
      bNxt[ct].q[0] = bp[0];
      bNxt[ct].q[1] = bp[1];
    }
#pragma unroll
    for (int ct = 0; ct < NCT; ++ct)
      acc[ct] = __builtin_amdgcn_wmma_f32_16x16x32_bf16(false, aCur.v, false, bCur[ct].v,
                                                        (short)0, acc[ct], false, false);
    aCur = aNxt;
#pragma unroll
    for (int ct = 0; ct < NCT; ++ct) bCur[ct] = bNxt[ct];
  }

  const int col0 = nBase + (lane & 15);
  const int rowS = mBase + (halfSel ? 8 : 0);
#pragma unroll
  for (int ct = 0; ct < NCT; ++ct) {
    int col = col0 + ct * 16;
    float bv = bias[col];
#pragma unroll
    for (int r = 0; r < 8; ++r) {
      float vv = acc[ct][r] + bv;
      if (do_elu) vv = (vv > 0.f) ? vv : expm1f(vv);
      out[(size_t)(rowS + r) * Nout + col] = vv;
    }
  }
}

// ---------------- edge phase (segment softmax attention) ----------------
__global__ void edge_score_kernel(const float* __restrict__ xl, const float* __restrict__ xr,
                                  const int* __restrict__ ei, const float* __restrict__ att,
                                  float* __restrict__ esc, unsigned* __restrict__ mmax,
                                  int etot, int nedges) {
  int t = blockIdx.x * blockDim.x + threadIdx.x;
  if (t >= etot * NHEAD) return;
  int e = t >> 3, h = t & 7;
  int s, d;
  if (e < nedges) { s = ei[e]; d = ei[nedges + e]; } else { s = d = e - nedges; }
  const float* pl = xl + (size_t)s * HCH + h * CCH;
  const float* pr = xr + (size_t)d * HCH + h * CCH;
  const float* pa = att + h * CCH;
  float acc = 0.f;
#pragma unroll
  for (int c = 0; c < CCH; ++c) {
    float z = pl[c] + pr[c];
    z = (z > 0.f) ? z : SLOPE * z;
    acc += pa[c] * z;
  }
  esc[t] = acc;
  atomicMax(&mmax[(size_t)d * NHEAD + h], fenc(acc));
}

__global__ void edge_psum_kernel(const int* __restrict__ ei, const unsigned* __restrict__ mmax,
                                 float* __restrict__ esc, float* __restrict__ ssum,
                                 int etot, int nedges) {
  int t = blockIdx.x * blockDim.x + threadIdx.x;
  if (t >= etot * NHEAD) return;
  int e = t >> 3, h = t & 7;
  int d = (e < nedges) ? ei[nedges + e] : (e - nedges);
  float m = fdec(mmax[(size_t)d * NHEAD + h]);
  float p = __expf(esc[t] - m);
  esc[t] = p;                                     // in-place: score -> exp weight
  atomicAdd(&ssum[(size_t)d * NHEAD + h], p);
}

__global__ void edge_agg_kernel(const float* __restrict__ xl, const int* __restrict__ ei,
                                const float* __restrict__ esc, const float* __restrict__ ssum,
                                float* __restrict__ out, int etot, int nedges) {
  int t = blockIdx.x * blockDim.x + threadIdx.x;
  if (t >= etot * NHEAD) return;
  int e = t >> 3, h = t & 7;
  int s, d;
  if (e < nedges) { s = ei[e]; d = ei[nedges + e]; } else { s = d = e - nedges; }
  float a = esc[t] / ssum[(size_t)d * NHEAD + h];
  const float* pl = xl + (size_t)s * HCH + h * CCH;
  __builtin_prefetch(pl, 0, 1);                   // global_prefetch: L2-resident gather
  float* po = out + (size_t)d * HCH + h * CCH;
#pragma unroll
  for (int c = 0; c < CCH; ++c) atomicAdd(&po[c], a * pl[c]);
}

// ---------------- final tiny classifier ----------------
__global__ void lin1_kernel(const float* __restrict__ h, const float* __restrict__ w,
                            const float* __restrict__ b, float* __restrict__ out, int n) {
  int t = blockIdx.x * blockDim.x + threadIdx.x;
  if (t >= n * NCLS) return;
  int i = t / NCLS, k = t - i * NCLS;
  float acc = b[k];
  const float* hr = h + (size_t)i * CCH;
#pragma unroll
  for (int c = 0; c < CCH; ++c) acc += hr[c] * w[c * NCLS + k];
  out[t] = acc;
}

// ---------------- host orchestration ----------------
extern "C" void kernel_launch(void* const* d_in, const int* in_sizes, int n_in,
                              void* d_out, int out_size, void* d_ws, size_t ws_size,
                              hipStream_t stream) {
  (void)in_sizes; (void)n_in; (void)out_size; (void)ws_size;

  const float* x     = (const float*)d_in[0];
  const int*   ei    = (const int*)  d_in[1];
  const float* n0g   = (const float*)d_in[2];  const float* n0b   = (const float*)d_in[3];
  const float* n1g   = (const float*)d_in[4];  const float* n1b   = (const float*)d_in[5];
  const float* n2g   = (const float*)d_in[6];  const float* n2b   = (const float*)d_in[7];
  const float* c0wl  = (const float*)d_in[8];  const float* c0bl  = (const float*)d_in[9];
  const float* c0wr  = (const float*)d_in[10]; const float* c0br  = (const float*)d_in[11];
  const float* c0att = (const float*)d_in[12]; const float* c0bias= (const float*)d_in[13];
  const float* cwl   = (const float*)d_in[14]; const float* cbl   = (const float*)d_in[15];
  const float* cwr   = (const float*)d_in[16]; const float* cbr   = (const float*)d_in[17];
  const float* catt  = (const float*)d_in[18]; const float* cbias = (const float*)d_in[19];
  const float* l0w   = (const float*)d_in[20]; const float* l0b   = (const float*)d_in[21];
  const float* l1w   = (const float*)d_in[22]; const float* l1b   = (const float*)d_in[23];

  char* ws = (char*)d_ws;
  size_t off = 0;
  auto take = [&](size_t bytes) -> char* {
    char* p = ws + off;
    off += (bytes + 255) & ~(size_t)255;
    return p;
  };
  float*          hA   = (float*)take((size_t)NNODES * HCH * 4);
  float*          xlg  = (float*)take((size_t)NNODES * HCH * 4);
  float*          xrg  = (float*)take((size_t)NNODES * HCH * 4);
  float*          agg  = (float*)take((size_t)NNODES * HCH * 4);
  float*          h32  = (float*)take((size_t)NNODES * CCH * 4);
  float*          mu   = (float*)take(1024);
  float*          inv  = (float*)take(1024);
  float*          sumb = (float*)take(1024);
  float*          sqb  = (float*)take(1024);
  unsigned short* abf  = (unsigned short*)take((size_t)NNODES * HCH * 2);
  unsigned short* wpL  = (unsigned short*)take((size_t)HCH * HCH * 2);
  unsigned short* wpR  = (unsigned short*)take((size_t)HCH * HCH * 2);
  float*          esc  = (float*)take((size_t)ETOTAL * NHEAD * 4);
  unsigned*       mmax = (unsigned*)take((size_t)NNODES * NHEAD * 4);
  float*          ssum = (float*)take((size_t)NNODES * NHEAD * 4);

  auto gs = [](long n) { return dim3((unsigned)((n + 255) / 256)); };

  auto run_bn_stats = [&](const float* xx, const float* rr, int f) {
    fill_f32_kernel<<<1, 256, 0, stream>>>(sumb, 0.f, 256);
    fill_f32_kernel<<<1, 256, 0, stream>>>(sqb, 0.f, 256);
    bn_partial_kernel<<<256, 256, 0, stream>>>(xx, rr, sumb, sqb, NNODES, f);
    bn_finalize_kernel<<<1, 256, 0, stream>>>(sumb, sqb, mu, inv, NNODES, f);
  };

  auto run_gemm = [&](const unsigned short* Ab, const unsigned short* Wpk,
                      const float* bias, float* outp, int K, int Nout, int elu) {
    dim3 g((Nout + 255) / 256, (NNODES + 31) / 32);
    if ((Nout & 63) == 0)
      gemm_bf16_wmma_kernel<4><<<g, 256, 0, stream>>>(Ab, Wpk, bias, outp,
                                                      NNODES, K, Nout, elu);
    else
      gemm_bf16_wmma_kernel<2><<<g, 256, 0, stream>>>(Ab, Wpk, bias, outp,
                                                      NNODES, K, Nout, elu);
  };

  auto run_edges = [&](const float* att, const float* bias) {
    fill_u32_kernel<<<gs((long)NNODES * NHEAD), 256, 0, stream>>>(mmax, 0u, NNODES * NHEAD);
    fill_f32_kernel<<<gs((long)NNODES * NHEAD), 256, 0, stream>>>(ssum, 0.f, NNODES * NHEAD);
    init_bias_kernel<<<gs((long)NNODES * HCH), 256, 0, stream>>>(agg, bias, NNODES * HCH, HCH);
    long tE = (long)ETOTAL * NHEAD;
    edge_score_kernel<<<gs(tE), 256, 0, stream>>>(xlg, xrg, ei, att, esc, mmax, ETOTAL, NEDGES);
    edge_psum_kernel <<<gs(tE), 256, 0, stream>>>(ei, mmax, esc, ssum, ETOTAL, NEDGES);
    edge_agg_kernel  <<<gs(tE), 256, 0, stream>>>(xlg, ei, esc, ssum, agg, ETOTAL, NEDGES);
  };

  // ---- h = BN(x) ; conv0 ----
  run_bn_stats(x, nullptr, NFEAT);
  bn_apply_kernel<<<gs((long)NNODES * NFEAT), 256, 0, stream>>>(
      x, nullptr, mu, inv, n0g, n0b, hA, NNODES * NFEAT, NFEAT, 0);
  to_bf16_kernel<<<gs((long)NNODES * NFEAT), 256, 0, stream>>>(hA, abf, NNODES * NFEAT);
  pack_w_kernel<<<gs((long)NFEAT * HCH), 256, 0, stream>>>(c0wl, wpL, NFEAT, HCH);
  pack_w_kernel<<<gs((long)NFEAT * HCH), 256, 0, stream>>>(c0wr, wpR, NFEAT, HCH);
  run_gemm(abf, wpL, c0bl, xlg, NFEAT, HCH, 0);
  run_gemm(abf, wpR, c0br, xrg, NFEAT, HCH, 0);
  run_edges(c0att, c0bias);

  // ---- h = BN1(conv0 out) ----
  run_bn_stats(agg, nullptr, HCH);
  bn_apply_kernel<<<gs((long)NNODES * HCH), 256, 0, stream>>>(
      agg, nullptr, mu, inv, n1g, n1b, hA, NNODES * HCH, HCH, 0);

  // ---- 3 residual GATv2 blocks: h = elu(BN2(gat(h) + h)) ----
  for (int i = 0; i < NCONVS; ++i) {
    to_bf16_kernel<<<gs((long)NNODES * HCH), 256, 0, stream>>>(hA, abf, NNODES * HCH);
    pack_w_kernel<<<gs((long)HCH * HCH), 256, 0, stream>>>(cwl + (size_t)i * HCH * HCH, wpL, HCH, HCH);
    pack_w_kernel<<<gs((long)HCH * HCH), 256, 0, stream>>>(cwr + (size_t)i * HCH * HCH, wpR, HCH, HCH);
    run_gemm(abf, wpL, cbl + (size_t)i * HCH, xlg, HCH, HCH, 0);
    run_gemm(abf, wpR, cbr + (size_t)i * HCH, xrg, HCH, HCH, 0);
    run_edges(catt + (size_t)i * NHEAD * CCH, cbias + (size_t)i * HCH);
    run_bn_stats(agg, hA, HCH);
    bn_apply_kernel<<<gs((long)NNODES * HCH), 256, 0, stream>>>(
        agg, hA, mu, inv, n2g, n2b, hA, NNODES * HCH, HCH, 1);
  }

  // ---- h = elu(h @ lin0 + b0) ; out = h @ lin1 + b1 ----
  to_bf16_kernel<<<gs((long)NNODES * HCH), 256, 0, stream>>>(hA, abf, NNODES * HCH);
  pack_w_kernel<<<gs((long)HCH * CCH), 256, 0, stream>>>(l0w, wpL, HCH, CCH);
  run_gemm(abf, wpL, l0b, h32, HCH, CCH, 1);
  lin1_kernel<<<gs((long)NNODES * NCLS), 256, 0, stream>>>(h32, l1w, l1b, (float*)d_out, NNODES);
}
